// GRIL_13408887898719
// MI455X (gfx1250) — matrix operand order repaired and checked by reading, not compiled, not run on hardware
//
#include <hip/hip_runtime.h>
#include <math.h>

// ---------------- types & helpers ----------------
typedef __attribute__((ext_vector_type(16))) __bf16         v16bf;
typedef __attribute__((ext_vector_type(8)))  float          v8f;
typedef __attribute__((ext_vector_type(8)))  unsigned short us8;

static __device__ __forceinline__ __bf16 f2bf(float f) {
  unsigned u = __builtin_bit_cast(unsigned, f);
  unsigned r = u + 0x7fffu + ((u >> 16) & 1u);           // RNE
  unsigned short h = (unsigned short)(r >> 16);
  return __builtin_bit_cast(__bf16, h);
}
static __device__ __forceinline__ float bf2f(__bf16 b) {
  unsigned short h = __builtin_bit_cast(unsigned short, b);
  unsigned u = ((unsigned)h) << 16;
  return __builtin_bit_cast(float, u);
}

union Frag { v16bf v; us8 h[2]; };

// A fragment (16x32 bf16): lanes 0-15 hold row M=lane, elems 0-7 = K0-7,
// elems 8-15 = K16-23; lanes 16-31 same row, K+8 / K+24.
static __device__ __forceinline__ v16bf ldA(const __bf16* p0, int lane) {
  int off = (lane & 16) ? 8 : 0;
  Frag f;
  f.h[0] = *(const us8*)(p0 + off);
  f.h[1] = *(const us8*)(p0 + 16 + off);
  return f.v;
}
// B fragment (32x16 bf16) from LDS, N-major storage (col*stride + k):
// lanes 0-15: elems 0-7 = K0-7, elems 8-15 = K16-23; lanes 16-31: K+8 / K+24.
static __device__ __forceinline__ v16bf ldB(const __bf16* p0, int lane) {
  const __bf16* p = p0 + ((lane & 16) ? 8 : 0);
  Frag f;
  f.h[0] = *(const us8*)(p);
  f.h[1] = *(const us8*)(p + 16);
  return f.v;
}
static __device__ __forceinline__ v8f wmma_bf16(v16bf a, v16bf b, v8f c) {
  return __builtin_amdgcn_wmma_f32_16x16x32_bf16(false, a, false, b, (short)0, c,
                                                 false, false);
}

// Async global->LDS 16B copy (per-lane), tracked by ASYNCcnt.
static __device__ __forceinline__ void async_copy_b128(unsigned lds_addr,
                                                       unsigned long long gaddr) {
  asm volatile("global_load_async_to_lds_b128 %0, %1, off"
               :: "v"(lds_addr), "v"(gaddr) : "memory");
}
static __device__ __forceinline__ void wait_asynccnt0() {
  asm volatile("s_wait_asynccnt 0x0" ::: "memory");
}

// ---------------- constants ----------------
#define BN   8192          // B * N
#define NB   512
#define HH   64
#define TT   32
#define EP_BIAS  0
#define EP_PRELU 1
#define EP_SIG   2
#define EP_TANH  3

// ================= GEMM over nodes: C[8192,64] = X[8192,K] * W[K,64] =================
// W stored o-major: W[o][K] (bf16). Staged fully into LDS via async-to-LDS DMA.
__global__ __launch_bounds__(256) void k_gemm_nodes(
    const __bf16* __restrict__ X, int ldx,
    const __bf16* __restrict__ W, const float* __restrict__ bias, int K,
    int ep, __bf16* __restrict__ outb, int ldob,
    float* __restrict__ outf, int t, const float* __restrict__ prelu_p)
{
  __shared__ __align__(16) __bf16 sW[64 * 480];
  __shared__ float sB[64];
  const int tid  = threadIdx.x;
  const int lane = tid & 31;
  const int wave = tid >> 5;

  // async DMA the weight panel into LDS (16B per lane per issue)
  {
    const unsigned ldsbase = (unsigned)(uintptr_t)(&sW[0]);
    const unsigned long long gbase = (unsigned long long)(uintptr_t)W;
    const int nchunk = (64 * K) >> 3;          // # of 16B chunks
    for (int i = tid; i < nchunk; i += 256)
      async_copy_b128(ldsbase + (unsigned)i * 16u, gbase + (size_t)i * 16u);
    wait_asynccnt0();
  }
  if (tid < 64) sB[tid] = bias[tid];
  __syncthreads();

  const int rowA = blockIdx.x * 128 + wave * 16 + (lane & 15);
  const __bf16* xrow = X + (size_t)rowA * ldx;

  v8f acc[4] = {};
  for (int kb = 0; kb < K; kb += 32) {
    v16bf a = ldA(xrow + kb, lane);
#pragma unroll
    for (int nt = 0; nt < 4; nt++) {
      v16bf b = ldB(sW + (size_t)(nt * 16 + (lane & 15)) * K + kb, lane);
      acc[nt] = wmma_bf16(a, b, acc[nt]);
    }
  }

  const int mBase = blockIdx.x * 128 + wave * 16 + ((lane & 16) ? 8 : 0);
  const int col0  = lane & 15;
  float pa = (ep == EP_PRELU) ? prelu_p[0] : 0.f;
#pragma unroll
  for (int nt = 0; nt < 4; nt++) {
    int col = nt * 16 + col0;
    float bv = sB[col];
#pragma unroll
    for (int r = 0; r < 8; r++) {
      int row = mBase + r;
      float v = acc[nt][r] + bv;
      if (ep == EP_BIAS) {
        outb[(size_t)row * ldob + col] = f2bf(v);
      } else if (ep == EP_PRELU) {
        float o = (v >= 0.f) ? v : pa * v;
        outb[(size_t)row * ldob + col] = f2bf(o);
        int b = row >> 9, n = row & 511;
        outf[(((size_t)(b * 128 + col)) * NB + n) * TT + t] = o;   // representations ch<64
      } else if (ep == EP_SIG) {
        outf[(size_t)row * 64 + col] = 1.f / (1.f + __expf(-v));
      } else {
        outf[(size_t)row * 64 + col] = tanhf(v);
      }
    }
  }
}

// ============ Graph GEMM: per batch b: Y[w,c] = sum_v A[w,v] * X[b,v,c] ============
// A: bf16 [512][512] row-major (one support). X sliced at channel xcoff, nch in {64,96}.
__global__ __launch_bounds__(256) void k_gemm_graph(
    const __bf16* __restrict__ A,
    const __bf16* __restrict__ X, int ldx, int xcoff, int nch,
    __bf16* __restrict__ Y, int ldy, int ycoff)
{
  __shared__ __align__(16) __bf16 sX[96 * 40];   // K-chunk (32) transposed, stride 40
  const int tid  = threadIdx.x;
  const int lane = tid & 31;
  const int wave = tid >> 5;
  const int b    = blockIdx.y;
  const int mBlk = blockIdx.x * 128;
  const int ntiles = nch >> 4;
  const int nchunk = nch >> 3;                   // us8 per X row

  const int rowA = mBlk + wave * 16 + (lane & 15);
  const __bf16* arow = A + (size_t)rowA * NB;

  v8f acc[6] = {};
  for (int kb = 0; kb < NB; kb += 32) {
    // prefetch next A K-chunk while we stage/transpose X
    if (kb + 32 < NB) __builtin_prefetch((const void*)(arow + kb + 32), 0, 0);
    for (int i = tid; i < 32 * nchunk; i += 256) {
      int vloc = i / nchunk;
      int cpos = (i - vloc * nchunk) * 8;
      us8 d = *(const us8*)(X + ((size_t)(b * NB + kb + vloc)) * ldx + xcoff + cpos);
#pragma unroll
      for (int j = 0; j < 8; j++)
        sX[(cpos + j) * 40 + vloc] = __builtin_bit_cast(__bf16, (unsigned short)d[j]);
    }
    __syncthreads();
    v16bf a = ldA(arow + kb, lane);
    for (int nt = 0; nt < ntiles; nt++) {
      v16bf bb = ldB(sX + (size_t)(nt * 16 + (lane & 15)) * 40, lane);
      acc[nt] = wmma_bf16(a, bb, acc[nt]);
    }
    __syncthreads();
  }

  const int mBase = mBlk + wave * 16 + ((lane & 16) ? 8 : 0);
  const int col0  = lane & 15;
  for (int nt = 0; nt < ntiles; nt++) {
    int col = ycoff + nt * 16 + col0;
#pragma unroll
    for (int r = 0; r < 8; r++) {
      int row = mBase + r;
      Y[((size_t)(b * NB + row)) * ldy + col] = f2bf(acc[nt][r]);
    }
  }
}

// ---------------- setup / elementwise kernels ----------------
__global__ void k_cvt_adj(const float* __restrict__ a, __bf16* __restrict__ o, int n) {
  int i = blockIdx.x * 256 + threadIdx.x;
  if (i < n) o[i] = f2bf(a[i]);
}

// W [64][nblk*blkReal] row-major -> Wt [64][Kpad] with per-block zero padding
__global__ void k_repack_w(const float* __restrict__ W, __bf16* __restrict__ Wt,
                           int Kpad, int blkPad, int blkReal) {
  int i = blockIdx.x * 256 + threadIdx.x;
  if (i >= 64 * Kpad) return;
  int o = i / Kpad, k = i - o * Kpad;
  int blk = k / blkPad, c = k - blk * blkPad;
  int Kreal = (Kpad / blkPad) * blkReal;
  float v = (c < blkReal) ? W[(size_t)o * Kreal + blk * blkReal + c] : 0.f;
  Wt[(size_t)o * Kpad + k] = f2bf(v);
}

__global__ void k_init_h(float* hf, __bf16* hb) {
  int i = blockIdx.x * 256 + threadIdx.x;
  if (i < BN * HH) { hf[i] = 0.f; hb[i] = f2bf(0.f); }
}

__global__ void k_stage1(int t, const float* __restrict__ x,
                         const unsigned char* __restrict__ mask,
                         const float* __restrict__ firstW, const float* __restrict__ firstB,
                         const float* __restrict__ h, const __bf16* __restrict__ hb,
                         float* __restrict__ predOut, float* __restrict__ x1buf,
                         __bf16* __restrict__ decxin)
{
  int tid = blockIdx.x * 256 + threadIdx.x;
  if (tid >= BN) return;
  float dot = firstB[0];
#pragma unroll 8
  for (int c = 0; c < HH; c++) dot += h[(size_t)tid * HH + c] * firstW[c];
  int xi = tid * TT + t;
  predOut[xi] = dot;
  bool m = mask[xi] != 0;
  float x1 = m ? x[xi] : dot;
  x1buf[tid] = x1;
  __bf16* row = decxin + (size_t)tid * 96;
  row[0] = f2bf(x1);
  row[1] = f2bf(m ? 1.f : 0.f);
  for (int c = 0; c < HH; c++) row[2 + c] = hb[(size_t)tid * HH + c];
  __bf16 zb = f2bf(0.f);
  for (int c = 66; c < 96; c++) row[c] = zb;
}

__global__ void k_pack_h(int t, const float* __restrict__ h, const __bf16* __restrict__ hb,
                         __bf16* __restrict__ dout1, __bf16* __restrict__ dout2,
                         float* __restrict__ reprOut)
{
  int tid = blockIdx.x * 256 + threadIdx.x;
  if (tid >= BN) return;
  int b = tid >> 9, n = tid & 511;
  for (int c = 0; c < HH; c++) {
    __bf16 v = hb[(size_t)tid * HH + c];
    dout1[(size_t)tid * 128 + 64 + c] = v;
    dout2[(size_t)tid * 128 + 64 + c] = v;
    reprOut[(((size_t)(b * 128 + 64 + c)) * NB + n) * TT + t] = h[(size_t)tid * HH + c];
  }
}

__global__ void k_read_x2(int t, const unsigned char* __restrict__ mask,
                          const float* __restrict__ readW, const float* __restrict__ readB,
                          const __bf16* __restrict__ dout2, const float* __restrict__ x1buf,
                          const __bf16* __restrict__ hb,
                          float* __restrict__ impOut, __bf16* __restrict__ gxg,
                          __bf16* __restrict__ gxc)
{
  int tid = blockIdx.x * 256 + threadIdx.x;
  if (tid >= BN) return;
  float s = readB[0];
#pragma unroll 8
  for (int c = 0; c < 128; c++) s += bf2f(dout2[(size_t)tid * 128 + c]) * readW[c];
  int xi = tid * TT + t;
  bool m = mask[xi] != 0;
  float x2 = m ? x1buf[tid] : s;
  impOut[xi] = x2;
  __bf16* g  = gxg + (size_t)tid * 480;
  __bf16* gc = gxc + (size_t)tid * 480;
  __bf16 xv = f2bf(x2), mv = f2bf(m ? 1.f : 0.f), zb = f2bf(0.f);
  g[0] = xv; g[1] = mv;
  gc[0] = xv; gc[1] = mv;
  for (int c = 0; c < HH; c++) g[2 + c] = hb[(size_t)tid * HH + c];
  for (int c = 66; c < 96; c++) { g[c] = zb; gc[c] = zb; }
}

__global__ void k_fill_xc(const float* __restrict__ r, const float* __restrict__ h,
                          __bf16* __restrict__ gxc) {
  int tid = blockIdx.x * 256 + threadIdx.x;
  if (tid >= BN) return;
  for (int c = 0; c < HH; c++)
    gxc[(size_t)tid * 480 + 2 + c] = f2bf(r[(size_t)tid * HH + c] * h[(size_t)tid * HH + c]);
}

__global__ void k_update_h(int t, const float* __restrict__ u, const float* __restrict__ cg,
                           float* __restrict__ h, __bf16* __restrict__ hb,
                           float* __restrict__ statesOut)
{
  int tid = blockIdx.x * 256 + threadIdx.x;
  if (tid >= BN) return;
  int b = tid >> 9, n = tid & 511;
  for (int c = 0; c < HH; c++) {
    float uu = u[(size_t)tid * HH + c];
    float cc = cg[(size_t)tid * HH + c];
    float hh = h[(size_t)tid * HH + c];
    float hn = uu * hh + (1.f - uu) * cc;
    h[(size_t)tid * HH + c]  = hn;
    hb[(size_t)tid * HH + c] = f2bf(hn);
    statesOut[(((size_t)(b * HH + c)) * NB + n) * TT + t] = hn;
  }
}

// ---------------- host launcher ----------------
extern "C" void kernel_launch(void* const* d_in, const int* in_sizes, int n_in,
                              void* d_out, int out_size, void* d_ws, size_t ws_size,
                              hipStream_t stream) {
  (void)in_sizes; (void)n_in; (void)out_size; (void)ws_size;
  // inputs (setup_inputs order, params dict flattened)
  const float* x_in      = (const float*)d_in[0];
  const float* adj_f     = (const float*)d_in[1];
  const unsigned char* mask = (const unsigned char*)d_in[2];
  const float* first_W   = (const float*)d_in[3];
  const float* first_b   = (const float*)d_in[4];
  const float* dec_in_W  = (const float*)d_in[5];
  const float* dec_in_b  = (const float*)d_in[6];
  const float* dec_gc_W  = (const float*)d_in[7];
  const float* dec_gc_b  = (const float*)d_in[8];
  const float* dec_out_W = (const float*)d_in[9];
  const float* dec_out_b = (const float*)d_in[10];
  const float* prelu_a   = (const float*)d_in[11];
  const float* read_W    = (const float*)d_in[12];
  const float* read_b    = (const float*)d_in[13];
  const float* r_W       = (const float*)d_in[14];
  const float* r_b       = (const float*)d_in[15];
  const float* u_W       = (const float*)d_in[16];
  const float* u_b       = (const float*)d_in[17];
  const float* c_W       = (const float*)d_in[18];
  const float* c_b       = (const float*)d_in[19];

  // outputs: imputations | predictions | representations | states
  float* out       = (float*)d_out;
  float* impOut    = out;
  float* predOut   = out + 262144;
  float* reprOut   = out + 524288;
  float* statesOut = out + 524288 + 33554432;

  // workspace carve-up
  char* ws = (char*)d_ws;
  size_t off = 0;
  auto take = [&](size_t bytes) { void* p = ws + off; off = (off + bytes + 255) & ~(size_t)255; return p; };
  __bf16* adjb    = (__bf16*)take(2 * 512 * 512 * 2);
  __bf16* Wdecin  = (__bf16*)take(64 * 96 * 2);
  __bf16* Wdecgc  = (__bf16*)take(64 * 128 * 2);
  __bf16* Wdecout = (__bf16*)take(64 * 128 * 2);
  __bf16* Wr      = (__bf16*)take(64 * 480 * 2);
  __bf16* Wu      = (__bf16*)take(64 * 480 * 2);
  __bf16* Wc      = (__bf16*)take(64 * 480 * 2);
  float*  hf      = (float*) take((size_t)BN * 64 * 4);
  __bf16* hb      = (__bf16*)take((size_t)BN * 64 * 2);
  __bf16* decxin  = (__bf16*)take((size_t)BN * 96 * 2);
  __bf16* decx64  = (__bf16*)take((size_t)BN * 64 * 2);
  __bf16* gcbuf   = (__bf16*)take((size_t)BN * 128 * 2);
  __bf16* dout1   = (__bf16*)take((size_t)BN * 128 * 2);
  __bf16* dout2   = (__bf16*)take((size_t)BN * 128 * 2);
  __bf16* gxg     = (__bf16*)take((size_t)BN * 480 * 2);
  __bf16* gxc     = (__bf16*)take((size_t)BN * 480 * 2);
  float*  rbuf    = (float*) take((size_t)BN * 64 * 4);
  float*  ubuf    = (float*) take((size_t)BN * 64 * 4);
  float*  cbuf    = (float*) take((size_t)BN * 64 * 4);
  float*  x1buf   = (float*) take((size_t)BN * 4);

  const __bf16* adjF = adjb;                 // forward support
  const __bf16* adjB = adjb + 512 * 512;     // backward support

  // ---- one-time prep (deterministic, replayed each call) ----
  k_cvt_adj<<<2048, 256, 0, stream>>>(adj_f, adjb, 2 * 512 * 512);
  k_repack_w<<<24, 256, 0, stream>>>(dec_in_W,  Wdecin,  96, 96, 66);
  k_repack_w<<<32, 256, 0, stream>>>(dec_gc_W,  Wdecgc,  128, 128, 128);
  k_repack_w<<<32, 256, 0, stream>>>(dec_out_W, Wdecout, 128, 128, 128);
  k_repack_w<<<120, 256, 0, stream>>>(r_W, Wr, 480, 96, 66);
  k_repack_w<<<120, 256, 0, stream>>>(u_W, Wu, 480, 96, 66);
  k_repack_w<<<120, 256, 0, stream>>>(c_W, Wc, 480, 96, 66);
  k_init_h<<<2048, 256, 0, stream>>>(hf, hb);

  dim3 gGraph(4, 16);   // M tiles x batch
  // ---- time scan ----
  for (int t = 0; t < TT; t++) {
    k_stage1<<<32, 256, 0, stream>>>(t, x_in, mask, first_W, first_b, hf, hb,
                                     predOut, x1buf, decxin);
    // decoder: dec_in conv (66->64)
    k_gemm_nodes<<<64, 256, 0, stream>>>(decxin, 96, Wdecin, dec_in_b, 96,
                                         EP_BIAS, decx64, 64, nullptr, t, nullptr);
    // decoder graph conv: gc = [Af*x, Ab*x]
    k_gemm_graph<<<gGraph, 256, 0, stream>>>(adjF, decx64, 64, 0, 64, gcbuf, 128, 0);
    k_gemm_graph<<<gGraph, 256, 0, stream>>>(adjB, decx64, 64, 0, 64, gcbuf, 128, 64);
    // dec_gc conv (128->64) -> dout1 ch0..63
    k_gemm_nodes<<<64, 256, 0, stream>>>(gcbuf, 128, Wdecgc, dec_gc_b, 128,
                                         EP_BIAS, dout1, 128, nullptr, t, nullptr);
    // fill h halves of dout1/dout2 + representations ch64..127
    k_pack_h<<<32, 256, 0, stream>>>(t, hf, hb, dout1, dout2, reprOut);
    // dec_out conv (128->64), PReLU, writes dout2 ch0..63 + representations ch0..63
    k_gemm_nodes<<<64, 256, 0, stream>>>(dout1, 128, Wdecout, dec_out_b, 128,
                                         EP_PRELU, dout2, 128, reprOut, t, prelu_a);
    // readout -> x2, imputations, build xg (gxg blk0) and xc skeleton
    k_read_x2<<<32, 256, 0, stream>>>(t, mask, read_W, read_b, dout2, x1buf, hb,
                                      impOut, gxg, gxc);
    // gconv order-2 on xg: blocks [Af, Af^2, Ab, Ab^2] at ch 96/192/288/384
    k_gemm_graph<<<gGraph, 256, 0, stream>>>(adjF, gxg, 480, 0,   96, gxg, 480, 96);
    k_gemm_graph<<<gGraph, 256, 0, stream>>>(adjF, gxg, 480, 96,  96, gxg, 480, 192);
    k_gemm_graph<<<gGraph, 256, 0, stream>>>(adjB, gxg, 480, 0,   96, gxg, 480, 288);
    k_gemm_graph<<<gGraph, 256, 0, stream>>>(adjB, gxg, 480, 288, 96, gxg, 480, 384);
    // r, u gates (330->64, padded to 480)
    k_gemm_nodes<<<64, 256, 0, stream>>>(gxg, 480, Wr, r_b, 480, EP_SIG,
                                         nullptr, 0, rbuf, t, nullptr);
    k_gemm_nodes<<<64, 256, 0, stream>>>(gxg, 480, Wu, u_b, 480, EP_SIG,
                                         nullptr, 0, ubuf, t, nullptr);
    // xc = [x2, mf, r*h]
    k_fill_xc<<<32, 256, 0, stream>>>(rbuf, hf, gxc);
    // gconv order-2 on xc
    k_gemm_graph<<<gGraph, 256, 0, stream>>>(adjF, gxc, 480, 0,   96, gxc, 480, 96);
    k_gemm_graph<<<gGraph, 256, 0, stream>>>(adjF, gxc, 480, 96,  96, gxc, 480, 192);
    k_gemm_graph<<<gGraph, 256, 0, stream>>>(adjB, gxc, 480, 0,   96, gxc, 480, 288);
    k_gemm_graph<<<gGraph, 256, 0, stream>>>(adjB, gxc, 480, 288, 96, gxc, 480, 384);
    // c gate
    k_gemm_nodes<<<64, 256, 0, stream>>>(gxc, 480, Wc, c_b, 480, EP_TANH,
                                         nullptr, 0, cbuf, t, nullptr);
    // h_new = u*h + (1-u)*c ; write states
    k_update_h<<<32, 256, 0, stream>>>(t, ubuf, cbuf, hf, hb, statesOut);
  }
}
